// MixtureConfounderPrior_21792664060394
// MI455X (gfx1250) — matrix operand order repaired and checked by reading, not compiled, not run on hardware
//
#include <hip/hip_runtime.h>
#include <math.h>

typedef __attribute__((ext_vector_type(16))) _Float16 v16h;
typedef __attribute__((ext_vector_type(8)))  float    v8f;

#define DDIM 1024
#define KC   32
#define CDIM 128
#define CFD  128
#define BT   1024
#define LVMIN (-2.3025850929940455f)
#define LVMAX (0.6931471805599453f)

// ---------------------------------------------------------------------------
// K1: cm = 0.9*code_momentum + 0.1*code_embed ;  code_mu = cm @ mu_w1[D:]
//     code_lv = cm @ lv_w[D:]     (all tiny: 32x128 @ 128x128)
// ---------------------------------------------------------------------------
__global__ void prep_codes(const float* __restrict__ ce,
                           const float* __restrict__ cmom,
                           const float* __restrict__ mu_w1,
                           const float* __restrict__ lv_w,
                           float* __restrict__ cm,
                           float* __restrict__ code_mu,
                           float* __restrict__ code_lv) {
  int tid = threadIdx.x;
  for (int i = tid; i < KC * CDIM; i += 256)
    cm[i] = 0.9f * cmom[i] + 0.1f * ce[i];
  __syncthreads();
  for (int o = tid; o < KC * CFD; o += 256) {
    int k = o >> 7, c = o & 127;
    float s_mu = 0.f, s_lv = 0.f;
    for (int e = 0; e < CDIM; ++e) {
      float v = cm[k * CDIM + e];
      s_mu += v * mu_w1[(DDIM + e) * CFD + c];
      s_lv += v * lv_w[(DDIM + e) * CFD + c];
    }
    code_mu[o] = s_mu;
    code_lv[o] = s_lv;
  }
}

// ---------------------------------------------------------------------------
// K2: fused projections of h (1024x1024):
//   n in [0,128)   -> P_mu  = h @ mu_w1[:D]
//   n in [128,256) -> P_lv  = h @ lv_w[:D]
//   n in [256,512) -> MIX1  = silu(h @ mw_w1 + mw_b1)
// One wave per 16x16 output tile, f16 WMMA, K-loop of 32 via LDS staging.
// ---------------------------------------------------------------------------
__global__ void proj_gemm(const float* __restrict__ h,
                          const float* __restrict__ mu_w1,
                          const float* __restrict__ lv_w,
                          const float* __restrict__ mw_w1,
                          const float* __restrict__ mw_b1,
                          float* __restrict__ P_mu,
                          float* __restrict__ P_lv,
                          float* __restrict__ MIX1) {
  __shared__ float As[16][33];   // padded stride: conflict-free gathers
  __shared__ float Bs[32][17];
  const int lane = threadIdx.x;
  const int m0 = blockIdx.x * 16;
  const int n0g = blockIdx.y * 16;

  const float* W; int ld, col0, mode;
  if (n0g < 128)      { W = mu_w1; ld = CFD; col0 = n0g;       mode = 0; }
  else if (n0g < 256) { W = lv_w;  ld = CFD; col0 = n0g - 128; mode = 1; }
  else                { W = mw_w1; ld = 256; col0 = n0g - 256; mode = 2; }

  const int mA = lane & 15;
  const int hi = lane >> 4;
  v8f acc = {0.f, 0.f, 0.f, 0.f, 0.f, 0.f, 0.f, 0.f};

  for (int ks = 0; ks < DDIM / 32; ++ks) {
    const int k0 = ks * 32;
    #pragma unroll 4
    for (int r = 0; r < 16; ++r)
      As[r][lane] = h[(m0 + r) * DDIM + k0 + lane];
    #pragma unroll 4
    for (int i = 0; i < 16; ++i) {
      int idx = i * 32 + lane;
      int kk = idx >> 4, nn = idx & 15;
      Bs[kk][nn] = W[(k0 + kk) * ld + col0 + nn];
    }
    __syncthreads();

    v16h a, b;
    #pragma unroll
    for (int j = 0; j < 16; ++j) {
      int v = j >> 1, s = j & 1;
      // 16-bit A 16x32 layout: VGPR v<4 -> K base 0, v>=4 -> K base 16;
      // lanes 16-31 add +8. (ISA 7.12.2)
      int ka = ((v >= 4) ? 16 : 0) + (hi ? 8 : 0) + ((v & 3) << 1) + s;
      a[j] = (_Float16)As[mA][ka];
      // 16-bit B 32x16: same K order as A (j<8: K=8*hi+j; j>=8: K=16+8*hi+j-8)
      b[j] = (_Float16)Bs[((j >> 3) << 4) + (hi ? 8 : 0) + (j & 7)][mA];
    }
    acc = __builtin_amdgcn_wmma_f32_16x16x32_f16(false, a, false, b,
                                                 (short)0, acc, false, false);
    __syncthreads();
  }

  #pragma unroll
  for (int r = 0; r < 8; ++r) {
    int m = m0 + r + hi * 8;          // C/D layout: VGPR r -> row r (+8 for hi)
    int n = col0 + mA;
    float val = acc[r];
    if (mode == 0)      P_mu[m * CFD + n] = val;
    else if (mode == 1) P_lv[m * CFD + n] = val;
    else {
      float x = val + mw_b1[n];
      MIX1[m * 256 + n] = x / (1.f + expf(-x));   // SiLU
    }
  }
}

// ---------------------------------------------------------------------------
// K3: mix logits (256->32 matvec per token) + wave32 softmax
// ---------------------------------------------------------------------------
__global__ void mix_softmax(const float* __restrict__ MIX1,
                            const float* __restrict__ mw_w2,
                            const float* __restrict__ mw_b2,
                            float* __restrict__ out_mix) {
  const int bt = blockIdx.x;
  const int k = threadIdx.x;           // 32 lanes = 32 codes
  float acc = mw_b2[k];
  for (int m = 0; m < 256; ++m)
    acc += MIX1[bt * 256 + m] * mw_w2[m * KC + k];
  float mx = acc;
  #pragma unroll
  for (int off = 16; off >= 1; off >>= 1)
    mx = fmaxf(mx, __shfl_xor(mx, off, 32));
  float e = expf(acc - mx);
  float s = e;
  #pragma unroll
  for (int off = 16; off >= 1; off >>= 1)
    s += __shfl_xor(s, off, 32);
  out_mix[bt * KC + k] = e / s;
}

// ---------------------------------------------------------------------------
// K4: per token: build mu_pre rows for all 32 codes, LayerNorm, 32x128x128
//     GEMM with mu_w2 via f16 WMMA, tanh/clip; logvar clip-add fused in.
// One block (2 waves) per token; mu_w2 staged once in LDS as f16.
// ---------------------------------------------------------------------------
__global__ void mu_lv_kernel(const float* __restrict__ P_mu,
                             const float* __restrict__ P_lv,
                             const float* __restrict__ code_mu,
                             const float* __restrict__ code_lv,
                             const float* __restrict__ mu_b1,
                             const float* __restrict__ ln_g,
                             const float* __restrict__ ln_b,
                             const float* __restrict__ mu_w2,
                             const float* __restrict__ mu_b2,
                             const float* __restrict__ lv_b,
                             float* __restrict__ out_mu,
                             float* __restrict__ out_lv) {
  __shared__ _Float16 w2h[CFD * CFD];       // 32 KB, shared B operand
  __shared__ float    xbuf[2][16][132];     // padded: conflict-free rows
  __shared__ _Float16 xln[2][16][136];      // padded f16 A operand
  const int bt = blockIdx.x;
  const int tid = threadIdx.x;
  const int w = tid >> 5;                   // wave id: code tile w*16..w*16+15
  const int lane = tid & 31;

  for (int i = tid; i < CFD * CFD; i += 64)
    w2h[i] = (_Float16)mu_w2[i];

  // mu_pre rows for this wave's 16 codes
  for (int idx = lane; idx < 16 * CFD; idx += 32) {
    int r = idx >> 7, c = idx & 127;
    int kk = w * 16 + r;
    xbuf[w][r][c] = P_mu[bt * CFD + c] + code_mu[kk * CFD + c] + mu_b1[c];
  }

  // logvar output: pure streaming clip-add (block-wide)
  for (int idx = tid; idx < KC * CFD; idx += 64) {
    int kk = idx >> 7, c = idx & 127;
    float v = P_lv[bt * CFD + c] + code_lv[kk * CFD + c] + lv_b[c];
    v = fminf(fmaxf(v, LVMIN), LVMAX);
    out_lv[(bt * KC + kk) * CFD + c] = v;
  }
  __syncthreads();

  // LayerNorm: 2 lanes per row, combine halves with shfl_xor(16)
  {
    const int row = lane & 15;
    const int c0 = (lane >> 4) * 64;
    float s = 0.f;
    for (int c = 0; c < 64; ++c) s += xbuf[w][row][c0 + c];
    s += __shfl_xor(s, 16, 32);
    float mean = s * (1.f / 128.f);
    float vs = 0.f;
    for (int c = 0; c < 64; ++c) {
      float d = xbuf[w][row][c0 + c] - mean;
      vs += d * d;
    }
    vs += __shfl_xor(vs, 16, 32);
    float rstd = rsqrtf(vs * (1.f / 128.f) + 1e-5f);
    for (int c = 0; c < 64; ++c) {
      int cc = c0 + c;
      xln[w][row][cc] =
          (_Float16)((xbuf[w][row][cc] - mean) * rstd * ln_g[cc] + ln_b[cc]);
    }
  }
  __syncthreads();

  const int mA = lane & 15;
  const int hi = lane >> 4;
  for (int nt = 0; nt < 8; ++nt) {
    v8f acc = {0.f, 0.f, 0.f, 0.f, 0.f, 0.f, 0.f, 0.f};
    #pragma unroll
    for (int ks = 0; ks < 4; ++ks) {
      v16h a, b;
      #pragma unroll
      for (int j = 0; j < 16; ++j) {
        int v = j >> 1, s2 = j & 1;
        int ka = ks * 32 + ((v >= 4) ? 16 : 0) + (hi ? 8 : 0) + ((v & 3) << 1) + s2;
        a[j] = xln[w][mA][ka];
        b[j] = w2h[(ks * 32 + ((j >> 3) << 4) + (hi ? 8 : 0) + (j & 7)) * CFD + nt * 16 + mA];
      }
      acc = __builtin_amdgcn_wmma_f32_16x16x32_f16(false, a, false, b,
                                                   (short)0, acc, false, false);
    }
    #pragma unroll
    for (int r = 0; r < 8; ++r) {
      int kk = w * 16 + r + hi * 8;
      int n = nt * 16 + mA;
      float y = tanhf(acc[r] + mu_b2[n]);
      y = fminf(fmaxf(y, -3.f), 3.f);
      out_mu[(bt * KC + kk) * CFD + n] = y;
    }
  }
}

// ---------------------------------------------------------------------------
extern "C" void kernel_launch(void* const* d_in, const int* in_sizes, int n_in,
                              void* d_out, int out_size, void* d_ws, size_t ws_size,
                              hipStream_t stream) {
  const float* h             = (const float*)d_in[0];
  const float* code_embed    = (const float*)d_in[1];
  const float* code_momentum = (const float*)d_in[2];
  const float* mu_w1         = (const float*)d_in[3];
  const float* mu_b1         = (const float*)d_in[4];
  const float* ln_g          = (const float*)d_in[5];
  const float* ln_b          = (const float*)d_in[6];
  const float* mu_w2         = (const float*)d_in[7];
  const float* mu_b2         = (const float*)d_in[8];
  const float* lv_w          = (const float*)d_in[9];
  const float* lv_b          = (const float*)d_in[10];
  const float* mw_w1         = (const float*)d_in[11];
  const float* mw_b1         = (const float*)d_in[12];
  const float* mw_w2         = (const float*)d_in[13];
  const float* mw_b2         = (const float*)d_in[14];

  float* ws      = (float*)d_ws;
  float* cm      = ws;                    // 32*128
  float* code_mu = ws + 4096;             // 32*128
  float* code_lv = ws + 8192;             // 32*128
  float* P_mu    = ws + 12288;            // 1024*128
  float* P_lv    = P_mu + 131072;         // 1024*128
  float* MIX1    = P_lv + 131072;         // 1024*256   (total ~2.05 MB)

  float* out     = (float*)d_out;
  float* out_mix = out;                   // (B,T,K)        32768
  float* out_mu  = out + 32768;           // (B,T,K,CF)  4194304
  float* out_lv  = out + 32768 + 4194304; // (B,T,K,CF)  4194304

  prep_codes<<<1, 256, 0, stream>>>(code_embed, code_momentum, mu_w1, lv_w,
                                    cm, code_mu, code_lv);
  proj_gemm<<<dim3(64, 32), 32, 0, stream>>>(h, mu_w1, lv_w, mw_w1, mw_b1,
                                             P_mu, P_lv, MIX1);
  mix_softmax<<<1024, 32, 0, stream>>>(MIX1, mw_w2, mw_b2, out_mix);
  mu_lv_kernel<<<1024, 64, 0, stream>>>(P_mu, P_lv, code_mu, code_lv, mu_b1,
                                        ln_g, ln_b, mu_w2, mu_b2, lv_b,
                                        out_mu, out_lv);
}
